// MultiHeadAttention_54357106098895
// MI455X (gfx1250) — matrix operand order repaired and checked by reading, not compiled, not run on hardware
//
#include <hip/hip_runtime.h>
#include <hip/hip_bf16.h>

typedef __bf16 bf16;
typedef __bf16 v16bf __attribute__((ext_vector_type(16)));
typedef __bf16 v8bf  __attribute__((ext_vector_type(8)));
typedef __bf16 v4bf  __attribute__((ext_vector_type(4)));
typedef float  v8f   __attribute__((ext_vector_type(8)));
typedef unsigned int u32x4 __attribute__((ext_vector_type(4)));
typedef int i32x4 __attribute__((ext_vector_type(4)));
typedef int i32x8 __attribute__((ext_vector_type(8)));

#define FDIM 512
#define NTOK 1024
#define BATCH 8
#define HEADS 8
#define HDIM 64

union V16U { v16bf v; v8bf h[2]; };

__device__ inline v8f wmma_bf16(v16bf a, v16bf b, v8f c) {
  return __builtin_amdgcn_wmma_f32_16x16x32_bf16(
      /*neg_a=*/false, a, /*neg_b=*/false, b,
      /*c_mod=*/(short)0, c, /*reuse_a=*/false, /*reuse_b=*/false);
}

// TDM: issue one tensor_load_to_lds for a 2D bf16 tile:
//   tile_dim0 = 64 elements (contiguous), tile_dim1 = 32 rows, row stride 512 elems.
__device__ inline void tdm_load_tile_32x64(unsigned int lds_addr, const bf16* gptr) {
  unsigned long long ga = (unsigned long long)(size_t)gptr;
  u32x4 g0;
  g0[0] = 1u;                                                   // count=1, user mode
  g0[1] = lds_addr;                                             // LDS byte address
  g0[2] = (unsigned int)ga;                                     // global_addr[31:0]
  g0[3] = (unsigned int)((ga >> 32) & 0x1FFFFFFull) | 0x80000000u; // addr[56:32] | type=2
  i32x8 g1;
  g1[0] = 0x00010000;            // workgroup_mask=0, data_size=1 (2 bytes)
  g1[1] = (int)(64u << 16);      // tensor_dim0 = 64   (bits 79:48, low half)
  g1[2] = (int)(32u << 16);      // tensor_dim1 = 32   (bits 111:80, low half)
  g1[3] = (int)(64u << 16);      // tile_dim0  = 64    (bits 127:112)
  g1[4] = 32;                    // tile_dim1  = 32, tile_dim2 = 0
  g1[5] = 512;                   // tensor_dim0_stride = 512 (bits 207:160, low 32)
  g1[6] = 0;                     // stride hi / tensor_dim1_stride lo
  g1[7] = 0;
  i32x4 gz = {0, 0, 0, 0};
#if defined(__clang_major__) && (__clang_major__ >= 23)
  i32x8 gz8 = {0, 0, 0, 0, 0, 0, 0, 0};
  __builtin_amdgcn_tensor_load_to_lds(g0, g1, gz, gz, gz8, 0);
#else
  __builtin_amdgcn_tensor_load_to_lds(g0, g1, gz, gz, 0);
#endif
}

// ---------------- small VALU prep kernels ----------------

__global__ void style_kernel(const float* __restrict__ s,
                             const float* __restrict__ aff_w,
                             const float* __restrict__ aff_b,
                             float* __restrict__ style) {
  int idx = blockIdx.x * blockDim.x + threadIdx.x;      // b*F + i
  int b = idx / FDIM, i = idx % FDIM;
  const float4* sb = (const float4*)(s + (size_t)b * FDIM);
  const float4* wr = (const float4*)(aff_w + (size_t)i * FDIM);
  float acc = aff_b[i];
  for (int k = 0; k < FDIM / 4; ++k) {
    float4 a = sb[k], w = wr[k];
    acc += a.x * w.x + a.y * w.y + a.z * w.z + a.w * w.w;
  }
  style[idx] = acc;
}

__global__ void demod_kernel(const float* __restrict__ w,
                             const float* __restrict__ style,
                             float* __restrict__ demod) {
  int idx = blockIdx.x * blockDim.x + threadIdx.x;      // b*F + o
  int b = idx / FDIM, o = idx % FDIM;
  const float4* wr = (const float4*)(w + (size_t)o * FDIM);
  const float4* st = (const float4*)(style + (size_t)b * FDIM);
  float acc = 0.f;
  for (int k = 0; k < FDIM / 4; ++k) {
    float4 a = wr[k], sv = st[k];
    float t0 = a.x * sv.x, t1 = a.y * sv.y, t2 = a.z * sv.z, t3 = a.w * sv.w;
    acc += t0 * t0 + t1 * t1 + t2 * t2 + t3 * t3;
  }
  demod[idx] = rsqrtf(acc + 1e-8f);
}

__global__ void modw_kernel(const float* __restrict__ w,
                            const float* __restrict__ style,
                            const float* __restrict__ demod,
                            bf16* __restrict__ wm) {
  size_t e = ((size_t)blockIdx.x * blockDim.x + threadIdx.x) * 4;
  int b = (int)(e / ((size_t)FDIM * FDIM));
  int rem = (int)(e % ((size_t)FDIM * FDIM));
  int o = rem / FDIM, i = rem % FDIM;
  float d = demod[b * FDIM + o];
  float4 wv = *(const float4*)(w + (size_t)o * FDIM + i);
  float4 sv = *(const float4*)(style + (size_t)b * FDIM + i);
  v4bf r;
  r[0] = (bf16)(wv.x * sv.x * d);
  r[1] = (bf16)(wv.y * sv.y * d);
  r[2] = (bf16)(wv.z * sv.z * d);
  r[3] = (bf16)(wv.w * sv.w * d);
  *(v4bf*)(wm + e) = r;
}

__global__ void cvt_bf16_kernel(const float* __restrict__ x, bf16* __restrict__ y) {
  size_t e = ((size_t)blockIdx.x * blockDim.x + threadIdx.x) * 4;
  float4 v = *(const float4*)(x + e);
  v4bf r;
  r[0] = (bf16)v.x; r[1] = (bf16)v.y; r[2] = (bf16)v.z; r[3] = (bf16)v.w;
  *(v4bf*)(y + e) = r;
}

// ---------------- WMMA GEMM:  C[b] = A[b] (NTOKxF) @ Wm[b]^T (FxF) ----------------
// 4 waves/block share a 32x512 bf16 A slab staged with async global->LDS copies;
// each wave computes a 32x32 C tile (4 v_wmma per 32-wide K step).
template <bool OUT_BF16>
__global__ void __launch_bounds__(128) gemm_wmma(const bf16* __restrict__ A,
                                                 const bf16* __restrict__ W,
                                                 void* __restrict__ outv) {
  const int tid = threadIdx.x;
  const int wave = tid >> 5;
  const int l = tid & 31, col = l & 15, sel = (l >> 4) & 1;
  const int o0 = blockIdx.x * 128 + wave * 32;
  const int n0 = blockIdx.y * 32;
  const int b  = blockIdx.z;

  __shared__ bf16 lA[32 * FDIM];   // 32 KB: rows n0..n0+31 (contiguous in memory)

  // async stage: 32 KB contiguous, 128 lanes x 16 x b128
  {
    unsigned int lbase = (unsigned int)(size_t)&lA[0];
    const char* g = (const char*)(A + (size_t)b * NTOK * FDIM + (size_t)n0 * FDIM);
#pragma unroll
    for (int i = 0; i < 16; ++i) {
      unsigned int lo = lbase + (unsigned int)(tid * 16 + i * 2048);
      const char* gp = g + tid * 16 + i * 2048;
      asm volatile("global_load_async_to_lds_b128 %0, %1, off"
                   :: "v"(lo), "v"(gp) : "memory");
    }
    asm volatile("s_wait_asynccnt 0" ::: "memory");
  }
  __syncthreads();

  const bf16* Wb = W + (size_t)b * FDIM * FDIM;
  v8f c00 = {}, c01 = {}, c10 = {}, c11 = {};

  const bf16* a0l = &lA[(size_t)col * FDIM + sel * 8];
  const bf16* a1l = a0l + (size_t)16 * FDIM;
  const bf16* b0base = Wb + (size_t)(o0 + col) * FDIM + sel * 8;
  const bf16* b1base = b0base + (size_t)16 * FDIM;

  for (int k0 = 0; k0 < FDIM; k0 += 32) {
    __builtin_prefetch((const void*)(b0base + k0 + 32), 0, 3);
    __builtin_prefetch((const void*)(b1base + k0 + 32), 0, 3);
    V16U a0, a1, b0, b1;
    a0.h[0] = *(const v8bf*)(a0l + k0);      a0.h[1] = *(const v8bf*)(a0l + k0 + 16);
    a1.h[0] = *(const v8bf*)(a1l + k0);      a1.h[1] = *(const v8bf*)(a1l + k0 + 16);
    b0.h[0] = *(const v8bf*)(b0base + k0);   b0.h[1] = *(const v8bf*)(b0base + k0 + 16);
    b1.h[0] = *(const v8bf*)(b1base + k0);   b1.h[1] = *(const v8bf*)(b1base + k0 + 16);
    c00 = wmma_bf16(a0.v, b0.v, c00);
    c01 = wmma_bf16(a0.v, b1.v, c01);
    c10 = wmma_bf16(a1.v, b0.v, c10);
    c11 = wmma_bf16(a1.v, b1.v, c11);
  }

  if (OUT_BF16) {
    bf16* O = (bf16*)outv + (size_t)b * NTOK * FDIM;
#pragma unroll
    for (int r = 0; r < 8; ++r) {
      int row0 = n0 + r + 8 * sel, row1 = row0 + 16;
      O[(size_t)row0 * FDIM + o0 + col]      = (bf16)c00[r];
      O[(size_t)row0 * FDIM + o0 + 16 + col] = (bf16)c01[r];
      O[(size_t)row1 * FDIM + o0 + col]      = (bf16)c10[r];
      O[(size_t)row1 * FDIM + o0 + 16 + col] = (bf16)c11[r];
    }
  } else {
    float* O = (float*)outv + (size_t)b * NTOK * FDIM;
#pragma unroll
    for (int r = 0; r < 8; ++r) {
      int row0 = n0 + r + 8 * sel, row1 = row0 + 16;
      O[(size_t)row0 * FDIM + o0 + col]      = c00[r];
      O[(size_t)row0 * FDIM + o0 + 16 + col] = c01[r];
      O[(size_t)row1 * FDIM + o0 + col]      = c10[r];
      O[(size_t)row1 * FDIM + o0 + 16 + col] = c11[r];
    }
  }
}

// ---------------- flash attention, q = k = v = kqv head slice ----------------
// one wave per (b, h, 16-query tile); 32x64 K/V blocks double-buffered in LDS
// via TDM tensor_load_to_lds (TENSORcnt-tracked).
__global__ void __launch_bounds__(32) attn_kernel(const bf16* __restrict__ kqv,
                                                  bf16* __restrict__ outb) {
  const int n0 = blockIdx.x * 16;
  const int h  = blockIdx.y;
  const int b  = blockIdx.z;
  const int l = threadIdx.x, col = l & 15, sel = l >> 4;

  __shared__ bf16 kv[2][32 * HDIM];   // double-buffered key/value block
  __shared__ bf16 pb[16 * 32];        // P tile transpose buffer

  const bf16* base = kqv + ((size_t)b * NTOK) * FDIM + h * HDIM;
  const unsigned int lkv[2] = { (unsigned int)(size_t)&kv[0][0],
                                (unsigned int)(size_t)&kv[1][0] };

  // Q as two A-tiles covering d = 0..31 and 32..63
  V16U qa0, qa1;
  {
    const bf16* qp = base + (size_t)(n0 + col) * FDIM;
    qa0.h[0] = *(const v8bf*)(qp + sel * 8);
    qa0.h[1] = *(const v8bf*)(qp + 16 + sel * 8);
    qa1.h[0] = *(const v8bf*)(qp + 32 + sel * 8);
    qa1.h[1] = *(const v8bf*)(qp + 48 + sel * 8);
  }

  v8f oacc[4] = {{}, {}, {}, {}};
  float m8[8], s8[8];
#pragma unroll
  for (int r = 0; r < 8; ++r) { m8[r] = -1e30f; s8[r] = 0.f; }

  // prime the TDM pipeline with key block 0
  tdm_load_tile_32x64(lkv[0], base);

  for (int kb = 0; kb < NTOK; kb += 32) {
    const int cur = (kb >> 5) & 1;
    // prefetch next key block into the other buffer (redundant refetch at tail)
    const int knxt = (kb + 32 < NTOK) ? (kb + 32) : kb;
    tdm_load_tile_32x64(lkv[cur ^ 1], base + (size_t)knxt * FDIM);
    __builtin_amdgcn_s_wait_tensorcnt(1);   // oldest (current buffer) complete
    asm volatile("" ::: "memory");

    const bf16* kvc = &kv[cur][0];

    // scores: two 16x16 tiles (keys 0-15 / 16-31), K-dim = d (64) = 2 chained WMMAs
    v8f sc[2];
#pragma unroll
    for (int s2 = 0; s2 < 2; ++s2) {
      V16U kb0, kb1;
      const bf16* kr = kvc + (s2 * 16 + col) * HDIM + sel * 8;
      kb0.h[0] = *(const v8bf*)(kr);       kb0.h[1] = *(const v8bf*)(kr + 16);
      kb1.h[0] = *(const v8bf*)(kr + 32);  kb1.h[1] = *(const v8bf*)(kr + 48);
      v8f t = {};
      t = wmma_bf16(qa0.v, kb0.v, t);
      t = wmma_bf16(qa1.v, kb1.v, t);
      sc[s2] = t * 0.125f;                 // 1/sqrt(64)
    }

    // online softmax (row r of C-tile -> 16-lane reductions, wave32 halves)
    float alpha[8];
#pragma unroll
    for (int r = 0; r < 8; ++r) {
      float t = fmaxf(sc[0][r], sc[1][r]);
#pragma unroll
      for (int m = 1; m < 16; m <<= 1) t = fmaxf(t, __shfl_xor(t, m, 32));
      float mnew = fmaxf(m8[r], t);
      float a = __expf(m8[r] - mnew);
      float p0 = __expf(sc[0][r] - mnew);
      float p1 = __expf(sc[1][r] - mnew);
      float rs = p0 + p1;
#pragma unroll
      for (int m = 1; m < 16; m <<= 1) rs += __shfl_xor(rs, m, 32);
      s8[r] = s8[r] * a + rs;
      m8[r] = mnew;
      alpha[r] = a;
      sc[0][r] = p0;
      sc[1][r] = p1;
    }
#pragma unroll
    for (int r = 0; r < 8; ++r) {
      oacc[0][r] *= alpha[r]; oacc[1][r] *= alpha[r];
      oacc[2][r] *= alpha[r]; oacc[3][r] *= alpha[r];
    }

    // transpose P (C-layout) -> A-layout via LDS
#pragma unroll
    for (int r = 0; r < 8; ++r) {
      int row = r + 8 * sel;
      pb[row * 32 + col]      = (bf16)sc[0][r];
      pb[row * 32 + 16 + col] = (bf16)sc[1][r];
    }
    __syncthreads();

    V16U pa;
    {
      const bf16* pr = &pb[col * 32 + sel * 16];
      pa.h[0] = *(const v8bf*)(pr);
      pa.h[1] = *(const v8bf*)(pr + 8);
    }

    // O(16x64) += P(16x32) @ V(32x64): 4 output tiles along d
#pragma unroll
    for (int t = 0; t < 4; ++t) {
      V16U vb;
#pragma unroll
      for (int j = 0; j < 16; ++j)
        vb.v[j] = kvc[(sel * 16 + j) * HDIM + t * 16 + col];
      oacc[t] = wmma_bf16(pa.v, vb.v, oacc[t]);
    }
    __syncthreads();   // pb reused next block
  }

  // epilogue: divide by row sums, store bf16
#pragma unroll
  for (int t = 0; t < 4; ++t) {
#pragma unroll
    for (int r = 0; r < 8; ++r) {
      int row = n0 + r + 8 * sel;
      outb[((size_t)b * NTOK + row) * FDIM + h * HDIM + t * 16 + col] =
          (bf16)(oacc[t][r] / s8[r]);
    }
  }
}

// ---------------- host launcher ----------------
extern "C" void kernel_launch(void* const* d_in, const int* in_sizes, int n_in,
                              void* d_out, int out_size, void* d_ws, size_t ws_size,
                              hipStream_t stream) {
  const float* x   = (const float*)d_in[0];
  const float* s   = (const float*)d_in[1];
  const float* kw  = (const float*)d_in[2];
  const float* kaw = (const float*)d_in[3];
  const float* kab = (const float*)d_in[4];
  const float* ow  = (const float*)d_in[5];
  const float* oaw = (const float*)d_in[6];
  const float* oab = (const float*)d_in[7];
  float* out = (float*)d_out;

  char* ws = (char*)d_ws;
  size_t off = 0;
  float* style_k = (float*)(ws + off); off += (size_t)BATCH * FDIM * 4;
  float* style_o = (float*)(ws + off); off += (size_t)BATCH * FDIM * 4;
  float* demod_k = (float*)(ws + off); off += (size_t)BATCH * FDIM * 4;
  float* demod_o = (float*)(ws + off); off += (size_t)BATCH * FDIM * 4;
  off = (off + 255) & ~(size_t)255;
  bf16* xbf = (bf16*)(ws + off); off += (size_t)BATCH * NTOK * FDIM * 2;
  bf16* wkm = (bf16*)(ws + off); off += (size_t)BATCH * FDIM * FDIM * 2;
  bf16* wom = (bf16*)(ws + off); off += (size_t)BATCH * FDIM * FDIM * 2;
  bf16* kqv = (bf16*)(ws + off); off += (size_t)BATCH * NTOK * FDIM * 2;
  bf16* att = xbf;  // xbf dead after GEMM1; reuse for attention output

  cvt_bf16_kernel<<<4096, 256, 0, stream>>>(x, xbf);
  style_kernel<<<16, 256, 0, stream>>>(s, kaw, kab, style_k);
  style_kernel<<<16, 256, 0, stream>>>(s, oaw, oab, style_o);
  demod_kernel<<<16, 256, 0, stream>>>(kw, style_k, demod_k);
  demod_kernel<<<16, 256, 0, stream>>>(ow, style_o, demod_o);
  modw_kernel<<<2048, 256, 0, stream>>>(kw, style_k, demod_k, wkm);
  modw_kernel<<<2048, 256, 0, stream>>>(ow, style_o, demod_o, wom);

  gemm_wmma<true><<<dim3(FDIM / 128, NTOK / 32, BATCH), 128, 0, stream>>>(xbf, wkm, (void*)kqv);
  attn_kernel<<<dim3(NTOK / 16, HEADS, BATCH), 32, 0, stream>>>(kqv, att);
  gemm_wmma<false><<<dim3(FDIM / 128, NTOK / 32, BATCH), 128, 0, stream>>>(att, wom, (void*)out);
}
